// DeepSeekSparseAttention_9457517985813
// MI455X (gfx1250) — matrix operand order repaired and checked by reading, not compiled, not run on hardware
//
#include <hip/hip_runtime.h>

// ---------------------------------------------------------------------------
// Types for CDNA5 WMMA (wave32): v_wmma_f32_16x16x32_f16
// ---------------------------------------------------------------------------
typedef _Float16 h16;
typedef __attribute__((ext_vector_type(16))) _Float16 v16h;
typedef __attribute__((ext_vector_type(8)))  float    v8f;
typedef __attribute__((ext_vector_type(4)))  _Float16 v4h;

union V16U { v16h v; v4h q[4]; uint4 u[2]; _Float16 h[16]; };

static __device__ inline v8f wmma32(v16h a, v16h b, v8f c) {
    // D = A(16x32 f16) * B(32x16 f16) + C(16x16 f32)
    return __builtin_amdgcn_wmma_f32_16x16x32_f16(
        /*neg_a=*/false, a, /*neg_b=*/false, b,
        /*c_mod=*/(short)0, c, /*reuse_a=*/false, /*reuse_b=*/false);
}

// A-fragment (16x32 MxK): lane m = lane&15 holds row m; hl = lane>>4 selects
// K 0..15 (hl=0) or 16..31 (hl=1), kbase = hl*8. 16B-aligned sources.
static __device__ inline v16h load_afrag(const h16* row, int kbase) {
    V16U u;
    u.u[0] = *(const uint4*)(row + 2 * kbase);        // K 16*hl +0..7
    u.u[1] = *(const uint4*)(row + 2 * kbase + 8);    // K 16*hl +8..15
    return u.v;
}

// B-fragment (32x16 KxN): lane c<16 holds column c for K=0..15; lane c+16 for
// K=16..31.  With the stationary matrix stored transposed (col-major rows),
// the 16 halfs per lane are contiguous.  16B-aligned variant:
static __device__ inline v16h load_bfrag16_a16(const h16* p) {
    V16U u;
    u.u[0] = ((const uint4*)p)[0];
    u.u[1] = ((const uint4*)p)[1];
    return u.v;
}
// 8B-aligned variant (padded LDS strides):
static __device__ inline v16h load_bfrag16_a8(const h16* p) {
    V16U u;
    u.q[0] = ((const v4h*)p)[0];
    u.q[1] = ((const v4h*)p)[1];
    u.q[2] = ((const v4h*)p)[2];
    u.q[3] = ((const v4h*)p)[3];
    return u.v;
}

// ---------------------------------------------------------------------------
// Problem constants (fixed by the reference)
// ---------------------------------------------------------------------------
#define BB   2
#define SS   2048
#define DM   1024
#define NH   16
#define DK   64
#define NQKV 3072
#define NIDX 256
#define MROWS (BB * SS)          // 4096
#define MASKW (SS / 32)          // 64 mask words per query row

// ---------------------------------------------------------------------------
// 0a) fp32 -> fp16 conversion (x)
// ---------------------------------------------------------------------------
__global__ __launch_bounds__(256) void cvt_f32_f16(const float* __restrict__ s,
                                                   h16* __restrict__ d, int n) {
    for (int i = blockIdx.x * blockDim.x + threadIdx.x; i < n;
         i += gridDim.x * blockDim.x)
        d[i] = (h16)s[i];
}

// ---------------------------------------------------------------------------
// 0b) fp32 [K,N] -> fp16 transposed [N,K] (weights), tiled through LDS
// ---------------------------------------------------------------------------
__global__ __launch_bounds__(256) void transpose_cvt(
    const float* __restrict__ src, h16* __restrict__ dst, int K, int N) {
    __shared__ float tile[32][33];
    int n0 = blockIdx.x * 32, k0 = blockIdx.y * 32;
    int tx = threadIdx.x, ty = threadIdx.y;          // 32 x 8
#pragma unroll
    for (int j = 0; j < 4; ++j)
        tile[ty + j * 8][tx] = src[(size_t)(k0 + ty + j * 8) * N + n0 + tx];
    __syncthreads();
#pragma unroll
    for (int j = 0; j < 4; ++j)
        dst[(size_t)(n0 + ty + j * 8) * K + k0 + tx] =
            (h16)tile[tx][ty + j * 8];
}

// ---------------------------------------------------------------------------
// 1) WMMA GEMM, TN form: C[M,N] = A[M,K] * Bt[N,K]^T
//    No LDS: every fragment is a contiguous per-lane global b128 load; the
//    weight working set lives in the 192MB L2 across row-blocks.
//    Software pipelined: B frags issued as a batch, next A frags preloaded
//    before the WMMA burst, so waits are staggered instead of full drains.
//    Block = 128 threads (4 waves); wave computes 32 rows x 64 cols.
// ---------------------------------------------------------------------------
__global__ __launch_bounds__(128) void gemm_tn_wmma(
    const h16* __restrict__ A, const h16* __restrict__ Bt,
    h16* __restrict__ Ch, float* __restrict__ Cf, int M, int N, int Kd) {
    const int tid = threadIdx.x;
    const int lane = tid & 31, w = tid >> 5;
    const int hl = lane >> 4, ln = lane & 15;
    const int n0 = blockIdx.x * 64;
    const int m0 = blockIdx.y * 128 + w * 32;
    const int kb = hl * 8;

    const h16* arow0 = A + (size_t)(m0 + ln) * Kd;
    const h16* arow1 = A + (size_t)(m0 + 16 + ln) * Kd;
    const h16* bp[4];
#pragma unroll
    for (int i = 0; i < 4; ++i)
        bp[i] = Bt + (size_t)(n0 + i * 16 + ln) * Kd + hl * 16;

    v8f acc[2][4];
    v8f z = {};
#pragma unroll
    for (int g = 0; g < 2; ++g)
        for (int i = 0; i < 4; ++i) acc[g][i] = z;

    // prologue: first A fragments in flight
    v16h a0 = load_afrag(arow0, kb);
    v16h a1 = load_afrag(arow1, kb);

    for (int k0 = 0; k0 < Kd; k0 += 32) {
        // batch-issue all 4 B fragments for this K step
        v16h bfr[4];
#pragma unroll
        for (int i = 0; i < 4; ++i)
            bfr[i] = load_bfrag16_a16(bp[i] + k0);

        // preload next iteration's A fragments (overlaps the WMMA burst)
        v16h a0n, a1n;
        if (k0 + 32 < Kd) {
            a0n = load_afrag(arow0 + k0 + 32, kb);
            a1n = load_afrag(arow1 + k0 + 32, kb);
            __builtin_prefetch(arow0 + k0 + 64, 0, 3);
            __builtin_prefetch(arow1 + k0 + 64, 0, 3);
        }

#pragma unroll
        for (int i = 0; i < 4; ++i) {
            acc[0][i] = wmma32(a0, bfr[i], acc[0][i]);
            acc[1][i] = wmma32(a1, bfr[i], acc[1][i]);
        }
        if (k0 + 32 < Kd) { a0 = a0n; a1 = a1n; }
    }

#pragma unroll
    for (int g = 0; g < 2; ++g)
        for (int i = 0; i < 4; ++i)
            for (int r = 0; r < 8; ++r) {
                int row = m0 + g * 16 + r + 8 * hl;
                int col = n0 + i * 16 + ln;
                float v = acc[g][i][r];
                if (Cf) Cf[(size_t)row * N + col] = v;
                else    Ch[(size_t)row * N + col] = (h16)v;
            }
}

// ---------------------------------------------------------------------------
// 2) RoPE + split qkv[B*S,3,H,64] -> Q/K [B,H,S,64], V transposed [B,H,64,S]
// ---------------------------------------------------------------------------
__global__ __launch_bounds__(64) void rope_split(
    const h16* __restrict__ qkvh, h16* __restrict__ Qh,
    h16* __restrict__ Kh, h16* __restrict__ Vt) {
    int bid = blockIdx.x;
    int h = bid & (NH - 1);
    int bs = bid >> 4;              // b*S + s
    int s = bs & (SS - 1);
    int b = bs >> 11;
    const h16* row = qkvh + (size_t)bs * NQKV;
    int t = threadIdx.x;
    size_t outbase = ((size_t)(b * NH + h) * SS + s) * DK;

    // V transposed: Vt[b,h,d,s]
    Vt[((size_t)(b * NH + h) * DK + t) * SS + s] = row[2 * DM + h * DK + t];

    if (t < 32) {
        float freq = __powf(10000.f, -(float)t / 32.f);
        float sn, cs;
        __sincosf((float)s * freq, &sn, &cs);
        const h16* qrow = row + h * DK;
        const h16* krow = row + DM + h * DK;
        float q1 = (float)qrow[2 * t], q2 = (float)qrow[2 * t + 1];
        Qh[outbase + 2 * t]     = (h16)(q1 * cs - q2 * sn);
        Qh[outbase + 2 * t + 1] = (h16)(q1 * sn + q2 * cs);
        float k1 = (float)krow[2 * t], k2 = (float)krow[2 * t + 1];
        Kh[outbase + 2 * t]     = (h16)(k1 * cs - k2 * sn);
        Kh[outbase + 2 * t + 1] = (h16)(k1 * sn + k2 * cs);
    }
}

// ---------------------------------------------------------------------------
// 3) Index scores: sc[b,q,k] = (iq[b,q,:] . ik[b,k,:]) / 8 for k <= q (WMMA)
//    All fragments are contiguous per-lane global loads; causal tiles only.
// ---------------------------------------------------------------------------
__global__ __launch_bounds__(128) void idx_scores_wmma(
    const h16* __restrict__ iqh, const h16* __restrict__ ikh,
    float* __restrict__ sc) {
    int b = blockIdx.z;
    int q0 = blockIdx.y * 16;
    int k0b = blockIdx.x * 64;
    if (k0b > q0 + 15) return;                 // fully above diagonal
    int tid = threadIdx.x, lane = tid & 31, w = tid >> 5;
    int hl = lane >> 4, ln = lane & 15;
    int k0 = k0b + w * 16;
    if (k0 > q0 + 15) return;                  // wave-uniform -> EXEC stays full

    const h16* arow = iqh + ((size_t)b * SS + q0 + ln) * NIDX;
    const h16* bcol = ikh + ((size_t)b * SS + k0 + ln) * NIDX + hl * 16;

    v8f c = {};
    for (int d0 = 0; d0 < NIDX; d0 += 32) {
        v16h a  = load_afrag(arow + d0, hl * 8);
        v16h bb = load_bfrag16_a16(bcol + d0);
        c = wmma32(a, bb, c);
    }
    float* out = sc + (size_t)b * SS * SS;
    for (int r = 0; r < 8; ++r) {
        int q = q0 + r + 8 * hl;
        int k = k0 + ln;
        if (k <= q) out[(size_t)q * SS + k] = c[r] * 0.125f;
    }
}

// ---------------------------------------------------------------------------
// 4) Exact top-k per query row via 4-pass radix select -> 32-bit mask words
// ---------------------------------------------------------------------------
static __device__ inline unsigned fkey(float f) {
    unsigned u = __float_as_uint(f);
    return (u & 0x80000000u) ? ~u : (u | 0x80000000u);   // monotone order
}

__global__ __launch_bounds__(256) void topk_select(
    const float* __restrict__ sc, unsigned* __restrict__ mask,
    const int* __restrict__ kptr) {
    int b = blockIdx.y, q = blockIdx.x;
    int K = *kptr;
    if (K > SS) K = SS;
    const float* row = sc + ((size_t)b * SS + q) * SS;
    unsigned* mrow = mask + ((size_t)b * SS + q) * MASKW;
    int n = q + 1;
    int tid = threadIdx.x;

    if (n <= K) {                  // everything causal is selected
        for (int wd = tid; wd < MASKW; wd += 256) {
            int kb = wd * 32;
            unsigned v;
            if (q >= kb + 31)      v = 0xffffffffu;
            else if (q < kb)       v = 0u;
            else                   v = (2u << (q - kb)) - 1u;
            mrow[wd] = v;
        }
        return;
    }

    __shared__ unsigned hist[256];
    __shared__ unsigned s_pref;
    __shared__ int s_need;
    if (tid == 0) { s_pref = 0u; s_need = K; }
    __syncthreads();

    unsigned pref = 0u;
    for (int pass = 0; pass < 4; ++pass) {
        int shift = 24 - pass * 8;
        hist[tid] = 0u;
        __syncthreads();
        for (int k2 = tid; k2 < n; k2 += 256) {
            unsigned u = fkey(row[k2]);
            bool ok = (pass == 0) || (((u ^ pref) >> (shift + 8)) == 0u);
            if (ok) atomicAdd(&hist[(u >> shift) & 255u], 1u);
        }
        __syncthreads();
        if (tid == 0) {
            unsigned cum = 0u;
            int d = 255;
            for (; d >= 0; --d) {
                unsigned c = hist[d];
                if (cum + c >= (unsigned)s_need) break;
                cum += c;
            }
            if (d < 0) d = 0;
            s_pref |= ((unsigned)d) << shift;
            s_need -= (int)cum;
        }
        __syncthreads();
        pref = s_pref;
        __syncthreads();
    }
    unsigned T = pref;             // key of the K-th largest (exact)

    for (int wd = tid; wd < MASKW; wd += 256) {
        unsigned v = 0u;
        int kb = wd * 32;
        int lim = q - kb + 1;
        if (lim > 32) lim = 32;
        for (int i = 0; i < lim; ++i)
            if (fkey(row[kb + i]) >= T) v |= (1u << i);
        mrow[wd] = v;
    }
}

// ---------------------------------------------------------------------------
// 5) Sparse flash attention: block = 4 waves, 64 queries.
//    K staged [32][72] (padded), V staged from transposed global [64][40],
//    so ALL WMMA fragment reads are contiguous vector ds loads.
// ---------------------------------------------------------------------------
#define KS_STRIDE 72
#define VS_STRIDE 40

__global__ __launch_bounds__(128) void sparse_flash_attn(
    const h16* __restrict__ Qh, const h16* __restrict__ Kh,
    const h16* __restrict__ Vt, const unsigned* __restrict__ mask,
    h16* __restrict__ attnh) {
    __shared__ h16 Ks[32 * KS_STRIDE];
    __shared__ h16 Vs[64 * VS_STRIDE];
    __shared__ h16 Ps[4][16 * 32];

    int b = blockIdx.z, h = blockIdx.y;
    int q0b = blockIdx.x * 64;
    int tid = threadIdx.x, lane = tid & 31, w = tid >> 5;
    int hl = lane >> 4, ln = lane & 15;
    int q0 = q0b + w * 16;

    const h16* Qb = Qh + (size_t)(b * NH + h) * SS * DK;
    const h16* Kb = Kh + (size_t)(b * NH + h) * SS * DK;
    const h16* Vb = Vt + (size_t)(b * NH + h) * DK * SS;   // [64][S]
    const unsigned* mbase = mask + (size_t)b * SS * MASKW;

    const h16* qrowp = Qb + (size_t)(q0 + ln) * DK;
    v16h aq0 = load_afrag(qrowp + 0,  hl * 8);   // dims 0..31
    v16h aq1 = load_afrag(qrowp + 32, hl * 8);   // dims 32..63

    float mr[8], lr[8];
    v8f acc[4];
    v8f z = {};
    for (int r = 0; r < 8; ++r) { mr[r] = -3e38f; lr[r] = 0.f; }
    for (int i = 0; i < 4; ++i) acc[i] = z;

    int kend = q0b + 64;   // causal upper bound for this block (<= S)
    for (int kb = 0; kb < kend; kb += 32) {
        __syncthreads();
        {   // stage K tile 32x64 -> Ks[32][72]
            int r = tid >> 2, cb = (tid & 3) * 16;
            const h16* src = Kb + (size_t)(kb + r) * DK + cb;
            *(uint4*)(Ks + r * KS_STRIDE + cb)     = ((const uint4*)src)[0];
            *(uint4*)(Ks + r * KS_STRIDE + cb + 8) = ((const uint4*)src)[1];
            if (kb + 32 < kend) __builtin_prefetch(src + 32 * DK, 0, 3);
        }
        {   // stage V tile 64(d) x 32(k) -> Vs[64][40] from transposed global
            int d = tid >> 1, cb = (tid & 1) * 16;
            const h16* src = Vb + (size_t)d * SS + kb + cb;
            h16* dst = Vs + d * VS_STRIDE + cb;
            *(v4h*)(dst + 0)  = *(const v4h*)(src + 0);
            *(v4h*)(dst + 4)  = *(const v4h*)(src + 4);
            *(v4h*)(dst + 8)  = *(const v4h*)(src + 8);
            *(v4h*)(dst + 12) = *(const v4h*)(src + 12);
            if (kb + 32 < kend) __builtin_prefetch(src + 32, 0, 3);
        }
        __syncthreads();

        // ---- scores for two 16-key sub-tiles (contiguous b128 ds reads) ----
        v8f s0 = {}, s1 = {};
        {
            const h16* kc0 = Ks + (size_t)ln * KS_STRIDE + hl * 16;
            s0 = wmma32(aq0, load_bfrag16_a16(kc0), s0);
            s0 = wmma32(aq1, load_bfrag16_a16(kc0 + 32), s0);
            const h16* kc1 = Ks + (size_t)(16 + ln) * KS_STRIDE + hl * 16;
            s1 = wmma32(aq0, load_bfrag16_a16(kc1), s1);
            s1 = wmma32(aq1, load_bfrag16_a16(kc1 + 32), s1);
        }

        // ---- mask + online softmax ----------------------------------------
        int wword = kb >> 5;
        float p0[8], p1[8], rescale[8];
        for (int r = 0; r < 8; ++r) {
            int q = q0 + r + 8 * hl;
            unsigned mw = mbase[(size_t)q * MASKW + wword];
            float v0 = s0[r] * 0.125f;           // 1/sqrt(64)
            float v1 = s1[r] * 0.125f;
            if (!((mw >> ln) & 1u))        v0 = -3e38f;
            if (!((mw >> (16 + ln)) & 1u)) v1 = -3e38f;
            float rm = fmaxf(v0, v1);
            for (int off = 1; off < 16; off <<= 1)
                rm = fmaxf(rm, __shfl_xor(rm, off, 16));
            float nmr = fmaxf(mr[r], rm);
            float e0 = __expf(v0 - nmr), e1 = __expf(v1 - nmr);
            float rs = e0 + e1;
            for (int off = 1; off < 16; off <<= 1)
                rs += __shfl_xor(rs, off, 16);
            float scl = __expf(mr[r] - nmr);
            lr[r] = lr[r] * scl + rs;
            mr[r] = nmr;
            p0[r] = e0; p1[r] = e1; rescale[r] = scl;
        }
        for (int i = 0; i < 4; ++i)
            for (int r = 0; r < 8; ++r) acc[i][r] *= rescale[r];

        // ---- transpose P (C layout -> A layout) through per-wave LDS -------
        h16* pw = Ps[w];
        for (int r = 0; r < 8; ++r) {
            int prow = r + 8 * hl;
            pw[prow * 32 + ln]      = (h16)p0[r];
            pw[prow * 32 + 16 + ln] = (h16)p1[r];
        }
        asm volatile("s_wait_dscnt 0" ::: "memory");   // wave-local LDS RAW
        v16h ap = load_afrag(pw + (size_t)ln * 32, hl * 8);

        // ---- O += P(16x32) * V(32x64): 4 WMMAs, contiguous V frags ---------
        for (int i = 0; i < 4; ++i) {
            const h16* vc = Vs + (size_t)(i * 16 + ln) * VS_STRIDE + hl * 16;
            acc[i] = wmma32(ap, load_bfrag16_a8(vc), acc[i]);
        }
    }

    // ---- epilogue: O /= l, write heads-concatenated [B,S,1024] ------------
    for (int i = 0; i < 4; ++i)
        for (int r = 0; r < 8; ++r) {
            int q = q0 + r + 8 * hl;
            float o = acc[i][r] / lr[r];
            attnh[((size_t)(b * SS + q)) * DM + h * DK + i * 16 + ln] = (h16)o;
        }
}

// ---------------------------------------------------------------------------
// Host-side launch
// ---------------------------------------------------------------------------
extern "C" void kernel_launch(void* const* d_in, const int* in_sizes, int n_in,
                              void* d_out, int out_size, void* d_ws, size_t ws_size,
                              hipStream_t stream) {
    const float* x     = (const float*)d_in[0];
    const float* W_qkv = (const float*)d_in[1];
    const float* W_o   = (const float*)d_in[2];
    const float* W_iq  = (const float*)d_in[3];
    const float* W_ik  = (const float*)d_in[4];
    const int*   kptr  = (const int*)d_in[5];
    float* out = (float*)d_out;

    // workspace bump allocator (256B aligned)
    char* base = (char*)d_ws;
    size_t off = 0;
    auto alloc = [&](size_t bytes) -> void* {
        void* p = base + off;
        off += (bytes + 255) & ~(size_t)255;
        return p;
    };
    h16*      xh     = (h16*)alloc((size_t)MROWS * DM * 2);
    h16*      WqkvT  = (h16*)alloc((size_t)DM * NQKV * 2);   // [3072][1024]
    h16*      WiqT   = (h16*)alloc((size_t)DM * NIDX * 2);   // [256][1024]
    h16*      WikT   = (h16*)alloc((size_t)DM * NIDX * 2);
    h16*      WoT    = (h16*)alloc((size_t)DM * DM * 2);     // [1024][1024]
    h16*      qkvh   = (h16*)alloc((size_t)MROWS * NQKV * 2);
    h16*      Qh     = (h16*)alloc((size_t)BB * NH * SS * DK * 2);
    h16*      Kh     = (h16*)alloc((size_t)BB * NH * SS * DK * 2);
    h16*      Vth    = (h16*)alloc((size_t)BB * NH * DK * SS * 2);
    h16*      iqh    = (h16*)alloc((size_t)MROWS * NIDX * 2);
    h16*      ikh    = (h16*)alloc((size_t)MROWS * NIDX * 2);
    float*    isc    = (float*)alloc((size_t)BB * SS * SS * 4);
    unsigned* maskw  = (unsigned*)alloc((size_t)BB * SS * MASKW * 4);
    h16*      attnh  = (h16*)alloc((size_t)MROWS * DM * 2);

    // 0) conversions / weight transposes
    {
        size_t n = (size_t)MROWS * DM;
        cvt_f32_f16<<<4096, 256, 0, stream>>>(x, xh, (int)n);
    }
    transpose_cvt<<<dim3(NQKV / 32, DM / 32), dim3(32, 8), 0, stream>>>(
        W_qkv, WqkvT, DM, NQKV);
    transpose_cvt<<<dim3(NIDX / 32, DM / 32), dim3(32, 8), 0, stream>>>(
        W_iq, WiqT, DM, NIDX);
    transpose_cvt<<<dim3(NIDX / 32, DM / 32), dim3(32, 8), 0, stream>>>(
        W_ik, WikT, DM, NIDX);
    transpose_cvt<<<dim3(DM / 32, DM / 32), dim3(32, 8), 0, stream>>>(
        W_o, WoT, DM, DM);

    // 1) projections (TN WMMA GEMMs, register-direct, pipelined)
    gemm_tn_wmma<<<dim3(NQKV / 64, MROWS / 128), 128, 0, stream>>>(
        xh, WqkvT, qkvh, nullptr, MROWS, NQKV, DM);
    gemm_tn_wmma<<<dim3(NIDX / 64, MROWS / 128), 128, 0, stream>>>(
        xh, WiqT, iqh, nullptr, MROWS, NIDX, DM);
    gemm_tn_wmma<<<dim3(NIDX / 64, MROWS / 128), 128, 0, stream>>>(
        xh, WikT, ikh, nullptr, MROWS, NIDX, DM);

    // 2) RoPE + split (V transposed)
    rope_split<<<BB * SS * NH, 64, 0, stream>>>(qkvh, Qh, Kh, Vth);

    // 3) index scores (causal tiles only)
    idx_scores_wmma<<<dim3(SS / 64, SS / 16, BB), 128, 0, stream>>>(iqh, ikh, isc);

    // 4) top-k -> mask bits
    topk_select<<<dim3(SS, BB), 256, 0, stream>>>(isc, maskw, kptr);

    // 5) sparse flash attention
    sparse_flash_attn<<<dim3(SS / 64, NH, BB), 128, 0, stream>>>(
        Qh, Kh, Vth, maskw, attnh);

    // 6) output projection (f32 out)
    gemm_tn_wmma<<<dim3(DM / 64, MROWS / 128), 128, 0, stream>>>(
        attnh, WoT, nullptr, out, MROWS, DM, DM);
    (void)in_sizes; (void)n_in; (void)out_size; (void)ws_size;
}
